// MoELayer_43473658970765
// MI455X (gfx1250) — matrix operand order repaired and checked by reading, not compiled, not run on hardware
//
#include <hip/hip_runtime.h>

// ---------------------------------------------------------------------------
// MoE (dense top-2) for MI455X / gfx1250, bf16 WMMA pipeline.
// Shapes: T=8192 tokens, D=1024, H=2048, E=8, K(top)=2.
// Compute-bound (~0.825 TFLOP); weights in bf16 (100MB) are L2-resident.
// ---------------------------------------------------------------------------

typedef __attribute__((ext_vector_type(16))) __bf16 v16bf;
typedef __attribute__((ext_vector_type(8)))  float  v8f;
typedef __attribute__((ext_vector_type(4)))  unsigned int u32x4;
typedef __attribute__((ext_vector_type(8)))  int          i32x8;
typedef __attribute__((ext_vector_type(4)))  int          i32x4;
typedef unsigned short u16;
typedef unsigned int   u32;

#define T_TOK 8192
#define DDIM  1024
#define HDIM  2048
#define NEXP  8

#if defined(__has_builtin)
#if __has_builtin(__builtin_amdgcn_tensor_load_to_lds) && \
    __has_builtin(__builtin_amdgcn_s_wait_tensorcnt)
#define HAVE_TDM 1
#endif
#endif

// ---------------------------------------------------------------------------
// fp32 -> bf16 conversion (grid-stride)
// ---------------------------------------------------------------------------
__global__ void f2bf_kernel(const float* __restrict__ src,
                            __bf16* __restrict__ dst, int n) {
  int stride = gridDim.x * blockDim.x;
  for (int i = blockIdx.x * blockDim.x + threadIdx.x; i < n; i += stride)
    dst[i] = (__bf16)src[i];
}

// ---------------------------------------------------------------------------
// Router: one wave32 per token. logits = x_row @ Wr [1024 x 8], softmax,
// top-2, normalized combine weights -> cw[t][e]; aux accumulators.
// aux[0..7] = sum of probs per expert, aux[8..15] = usage counts.
// ---------------------------------------------------------------------------
__global__ __launch_bounds__(256) void router_kernel(
    const float* __restrict__ x, const float* __restrict__ Wr,
    float* __restrict__ cw, float* __restrict__ aux) {
  const int lane = threadIdx.x & 31;
  const int wID  = threadIdx.x >> 5;
  const int t    = blockIdx.x * 8 + wID;

  float a[8];
#pragma unroll
  for (int e = 0; e < 8; ++e) a[e] = 0.0f;

  const float* xr = x + (size_t)t * DDIM;
  for (int j = lane; j < DDIM; j += 32) {
    float xv = xr[j];
    const float4* wr4 = reinterpret_cast<const float4*>(Wr + j * NEXP);
    float4 w0 = wr4[0], w1 = wr4[1];
    a[0] += xv * w0.x; a[1] += xv * w0.y; a[2] += xv * w0.z; a[3] += xv * w0.w;
    a[4] += xv * w1.x; a[5] += xv * w1.y; a[6] += xv * w1.z; a[7] += xv * w1.w;
  }
#pragma unroll
  for (int off = 16; off > 0; off >>= 1)
#pragma unroll
    for (int e = 0; e < 8; ++e) a[e] += __shfl_xor(a[e], off);

  if (lane == 0) {
    float m = a[0];
#pragma unroll
    for (int e = 1; e < 8; ++e) m = fmaxf(m, a[e]);
    float p[8], s = 0.0f;
#pragma unroll
    for (int e = 0; e < 8; ++e) { p[e] = __expf(a[e] - m); s += p[e]; }
    float inv = 1.0f / s;
#pragma unroll
    for (int e = 0; e < 8; ++e) p[e] *= inv;
    int i1 = 0;
#pragma unroll
    for (int e = 1; e < 8; ++e) if (p[e] > p[i1]) i1 = e;
    int i2 = (i1 == 0) ? 1 : 0;
#pragma unroll
    for (int e = 0; e < 8; ++e) if (e != i1 && p[e] > p[i2]) i2 = e;
    float v1 = p[i1], v2 = p[i2];
    float invw = 1.0f / (v1 + v2);
    float row[8];
#pragma unroll
    for (int e = 0; e < 8; ++e) row[e] = 0.0f;
    row[i1] = v1 * invw;
    row[i2] = v2 * invw;
#pragma unroll
    for (int e = 0; e < 8; ++e) cw[(size_t)t * NEXP + e] = row[e];
#pragma unroll
    for (int e = 0; e < 8; ++e) atomicAdd(&aux[e], p[e]);
    atomicAdd(&aux[8 + i1], 1.0f);
    atomicAdd(&aux[8 + i2], 1.0f);
  }
}

// ---------------------------------------------------------------------------
// WMMA fragment helpers (bf16 16x16x32, wave32 layouts per CDNA5 ISA 7.12.2)
// ---------------------------------------------------------------------------
__device__ __forceinline__ v16bf load_a_frag(const __bf16* base, int rowStride,
                                             int mBase, int kBase, int lane) {
  int m = mBase + (lane & 15);
  int k = kBase + ((lane & 16) ? 8 : 0);
  const __bf16* p = base + (size_t)m * rowStride + k;
  union { v16bf v; uint4 q[2]; } u;
  u.q[0] = *reinterpret_cast<const uint4*>(p);
  u.q[1] = *reinterpret_cast<const uint4*>(p + 16);
  return u.v;
}

__device__ __forceinline__ v16bf load_b_fragT(const __bf16* baseT, int kStride,
                                              int nBase, int kBase, int lane) {
  int n = nBase + (lane & 15);
  int k = kBase + ((lane & 16) ? 8 : 0);
  const __bf16* p = baseT + (size_t)n * kStride + k;
  union { v16bf v; uint4 q[2]; } u;
  u.q[0] = *reinterpret_cast<const uint4*>(p);
  u.q[1] = *reinterpret_cast<const uint4*>(p + 16);
  return u.v;
}

__device__ __forceinline__ v8f wmma_bf16(v16bf a, v16bf b, v8f c) {
  return __builtin_amdgcn_wmma_f32_16x16x32_bf16(false, a, false, b,
                                                 (short)0, c, false, false);
}

__device__ __forceinline__ u32 pack2(u16 lo, u16 hi) {
  return (u32)lo | ((u32)hi << 16);
}

// ---------------------------------------------------------------------------
// Fused MoE FFN. Block = 256 threads (8 waves), owns 32 tokens x D=1024 out.
// Per wave: 128 output columns -> 2x8 f32 accumulator tiles (128 VGPRs).
// LDS (dynamic, 272,384 B):
//   X[32][1024] bf16 | W1T[32][1024] | W3T[32][1024] | W2T[1024][32]
//   S1[32][32] f32   | S3[32][32] f32 | Hc[32][32] bf16
// X tile is fetched by the Tensor Data Mover (tensor_load_to_lds).
// ---------------------------------------------------------------------------
__global__ __launch_bounds__(256) void moe_ffn_kernel(
    const __bf16* __restrict__ xb, const __bf16* __restrict__ w1b,
    const __bf16* __restrict__ w3b, const __bf16* __restrict__ w2b,
    const float* __restrict__ cw, float* __restrict__ out) {
  extern __shared__ char smem[];
  __bf16* ldsX  = (__bf16*)(smem);
  __bf16* ldsW1 = (__bf16*)(smem + 65536);
  __bf16* ldsW3 = (__bf16*)(smem + 131072);
  __bf16* ldsW2 = (__bf16*)(smem + 196608);
  float*  ldsS1 = (float*)(smem + 262144);
  float*  ldsS3 = (float*)(smem + 266240);
  __bf16* ldsH  = (__bf16*)(smem + 270336);

  const int tid  = threadIdx.x;
  const int lane = tid & 31;
  const int wID  = tid >> 5;
  const int t0   = blockIdx.x * 32;
  const int hi8  = (lane & 16) ? 8 : 0;

  // ---- X tile [32 x 1024] bf16 into LDS, row-major.
#ifdef HAVE_TDM
  if (wID == 0) {
    unsigned long long ga =
        (unsigned long long)(const void*)(xb + (size_t)t0 * DDIM);
    u32 ldsAddr = (u32)(unsigned long long)(void*)ldsX;
    u32x4 g0;
    g0[0] = 1u;                               // count=1 (user descriptor)
    g0[1] = ldsAddr;                          // lds_addr
    g0[2] = (u32)(ga & 0xFFFFFFFFu);          // global_addr[31:0]
    g0[3] = (u32)((ga >> 32) & 0x01FFFFFFu) | 0x80000000u;  // [56:32]|type=2
    i32x8 g1;
    g1[0] = (int)(1u << 16);                  // data_size=1 (2 bytes/elem)
    g1[1] = (int)((u32)(DDIM & 0xFFFF) << 16);   // tensor_dim0 lo16 @[63:48]
    g1[2] = (int)((u32)32 << 16);             // dim0 hi=0 | tensor_dim1=32
    g1[3] = (int)((u32)(DDIM & 0xFFFF) << 16);   // dim1 hi=0 | tile_dim0=1024
    g1[4] = (int)32;                          // tile_dim1=32, tile_dim2=0
    g1[5] = (int)DDIM;                        // tensor_dim0_stride=1024
    g1[6] = 0;
    g1[7] = 0;
    i32x4 z4 = {0, 0, 0, 0};
#if __clang_major__ >= 23
    i32x8 z8 = {0, 0, 0, 0, 0, 0, 0, 0};
    __builtin_amdgcn_tensor_load_to_lds(g0, g1, z4, z4, z8, 0);
#else
    __builtin_amdgcn_tensor_load_to_lds(g0, g1, z4, z4, 0);
#endif
    __builtin_amdgcn_s_wait_tensorcnt(0);
  }
#else
  for (int i = tid; i < 32 * 128; i += 256) {
    int r = i >> 7, g = (i & 127) << 3;
    *(uint4*)(ldsX + r * DDIM + g) =
        *(const uint4*)(xb + (size_t)(t0 + r) * DDIM + g);
  }
#endif

  v8f acc[2][8];
  v8f zero = {};
#pragma unroll
  for (int mi = 0; mi < 2; ++mi)
#pragma unroll
    for (int ni = 0; ni < 8; ++ni) acc[mi][ni] = zero;

  for (int e = 0; e < NEXP; ++e) {
    const __bf16* w1e = w1b + (size_t)e * DDIM * HDIM;
    const __bf16* w3e = w3b + (size_t)e * DDIM * HDIM;
    const __bf16* w2e = w2b + (size_t)e * HDIM * DDIM;

    for (int hc = 0; hc < HDIM; hc += 32) {
      __syncthreads();  // protect LDS chunk buffers from previous stage-2

      // --- w1/w3 chunk [1024 x 32] -> transposed [32][1024] in LDS.
      //     4-row register pack -> ds_store_b64 (no sub-word LDS traffic).
      for (int i = tid; i < 1024; i += 256) {
        int d0 = (i >> 2) << 2;      // row group of 4
        int g  = (i & 3) << 3;       // col group of 8
        union { uint4 u; u16 s[8]; } a[4], b[4];
#pragma unroll
        for (int dd = 0; dd < 4; ++dd) {
          a[dd].u = *(const uint4*)(w1e + (size_t)(d0 + dd) * HDIM + hc + g);
          b[dd].u = *(const uint4*)(w3e + (size_t)(d0 + dd) * HDIM + hc + g);
        }
#pragma unroll
        for (int j = 0; j < 8; ++j) {
          uint2 pa, pb;
          pa.x = pack2(a[0].s[j], a[1].s[j]);
          pa.y = pack2(a[2].s[j], a[3].s[j]);
          pb.x = pack2(b[0].s[j], b[1].s[j]);
          pb.y = pack2(b[2].s[j], b[3].s[j]);
          *(uint2*)((u16*)ldsW1 + (size_t)(g + j) * DDIM + d0) = pa;
          *(uint2*)((u16*)ldsW3 + (size_t)(g + j) * DDIM + d0) = pb;
        }
      }
      // --- w2 chunk [32 x 1024] -> transposed [1024][32] in LDS.
      //     Lanes read consecutive 16B (coalesced); b64 stores along K.
      for (int i = tid; i < 1024; i += 256) {
        int k0 = (i >> 7) << 2;      // k group of 4
        int g8 = (i & 127) << 3;     // col group of 8
        union { uint4 u; u16 s[8]; } a[4];
#pragma unroll
        for (int kk = 0; kk < 4; ++kk)
          a[kk].u = *(const uint4*)(w2e + (size_t)(hc + k0 + kk) * DDIM + g8);
#pragma unroll
        for (int j = 0; j < 8; ++j) {
          uint2 p;
          p.x = pack2(a[0].s[j], a[1].s[j]);
          p.y = pack2(a[2].s[j], a[3].s[j]);
          *(uint2*)((u16*)ldsW2 + (size_t)(g8 + j) * 32 + k0) = p;
        }
      }
      if (hc + 32 < HDIM)
        __builtin_prefetch(w1e + hc + 32, 0, 1);  // global_prefetch_b8
      __syncthreads();

      // --- Stage 1: waves 0-3 compute X@W1 tiles, waves 4-7 X@W3 tiles.
      {
        const int tI = wID & 3;
        const int tM = (tI >> 1) << 4;
        const int tN = (tI & 1) << 4;
        const __bf16* bsrc = (wID < 4) ? ldsW1 : ldsW3;
        v8f p = zero;
        for (int kk = 0; kk < DDIM; kk += 32) {
          v16bf af = load_a_frag(ldsX, DDIM, tM, kk, lane);
          v16bf bf = load_b_fragT(bsrc, DDIM, tN, kk, lane);
          p = wmma_bf16(af, bf, p);
        }
        float* sd = (wID < 4) ? ldsS1 : ldsS3;
#pragma unroll
        for (int v = 0; v < 8; ++v)
          sd[(tM + v + hi8) * 32 + tN + (lane & 15)] = p[v];
      }
      __syncthreads();

      // --- h = silu(s1) * s3 * cw[t,e]  (fold combine weight into rows)
      for (int i = tid; i < 1024; i += 256) {
        int m = i >> 5;
        float s1 = ldsS1[i], s3 = ldsS3[i];
        float g = (s1 / (1.0f + __expf(-s1))) * s3 *
                  cw[(size_t)(t0 + m) * NEXP + e];
        ldsH[i] = (__bf16)g;
      }
      __syncthreads();

      // --- Stage 2: acc += h[32x32] @ W2chunk[32 x (wave's 128 cols)]
      {
        v16bf a0 = load_a_frag(ldsH, 32, 0, 0, lane);
        v16bf a1 = load_a_frag(ldsH, 32, 16, 0, lane);
#pragma unroll
        for (int ni = 0; ni < 8; ++ni) {
          v16bf bf = load_b_fragT(ldsW2, 32, wID * 128 + ni * 16, 0, lane);
          acc[0][ni] = wmma_bf16(a0, bf, acc[0][ni]);
          acc[1][ni] = wmma_bf16(a1, bf, acc[1][ni]);
        }
      }
    }
  }

  // --- Write out: f32 C/D layout (lane<16: M=v, N=lane; lane>=16: M=v+8).
#pragma unroll
  for (int mi = 0; mi < 2; ++mi)
#pragma unroll
    for (int ni = 0; ni < 8; ++ni)
#pragma unroll
      for (int v = 0; v < 8; ++v)
        out[(size_t)(t0 + mi * 16 + v + hi8) * DDIM + wID * 128 + ni * 16 +
            (lane & 15)] = acc[mi][ni][v];
}

// ---------------------------------------------------------------------------
// aux_loss = E * sum_e (usage_mean[e] * prob_mean[e])
// ---------------------------------------------------------------------------
__global__ void aux_finalize_kernel(const float* __restrict__ aux,
                                    float* __restrict__ out) {
  if (threadIdx.x == 0 && blockIdx.x == 0) {
    float s = 0.0f;
#pragma unroll
    for (int e = 0; e < 8; ++e)
      s += (aux[8 + e] / (float)T_TOK) * (aux[e] / (float)T_TOK);
    out[(size_t)T_TOK * DDIM] = (float)NEXP * s;
  }
}

// ---------------------------------------------------------------------------
// Launch. Input dict order: x, Wr, w1, w2, w3  (note: w2 before w3!)
// ---------------------------------------------------------------------------
extern "C" void kernel_launch(void* const* d_in, const int* in_sizes, int n_in,
                              void* d_out, int out_size, void* d_ws,
                              size_t ws_size, hipStream_t stream) {
  const float* x  = (const float*)d_in[0];
  const float* Wr = (const float*)d_in[1];
  const float* w1 = (const float*)d_in[2];
  const float* w2 = (const float*)d_in[3];
  const float* w3 = (const float*)d_in[4];
  float* out = (float*)d_out;

  char* ws = (char*)d_ws;
  __bf16* xb  = (__bf16*)(ws + 0);           // 16,777,216 B
  __bf16* w1b = (__bf16*)(ws + 16777216);    // 33,554,432 B
  __bf16* w3b = (__bf16*)(ws + 50331648);    // 33,554,432 B
  __bf16* w2b = (__bf16*)(ws + 83886080);    // 33,554,432 B
  float*  cw  = (float*)(ws + 117440512);    //    262,144 B
  float*  aux = (float*)(ws + 117702656);    //         64 B

  hipMemsetAsync(aux, 0, 64, stream);
  f2bf_kernel<<<2048, 256, 0, stream>>>(x,  xb,  T_TOK * DDIM);
  f2bf_kernel<<<2048, 256, 0, stream>>>(w1, w1b, NEXP * DDIM * HDIM);
  f2bf_kernel<<<2048, 256, 0, stream>>>(w3, w3b, NEXP * DDIM * HDIM);
  f2bf_kernel<<<2048, 256, 0, stream>>>(w2, w2b, NEXP * HDIM * DDIM);
  router_kernel<<<T_TOK / 8, 256, 0, stream>>>(x, Wr, cw, aux);
  moe_ffn_kernel<<<T_TOK / 32, 256, 272384, stream>>>(xb, w1b, w3b, w2b, cw,
                                                      out);
  aux_finalize_kernel<<<1, 32, 0, stream>>>(aux, out);
}
